// MyLSTM_66786741452995
// MI455X (gfx1250) — matrix operand order repaired and checked by reading, not compiled, not run on hardware
//
#include <hip/hip_runtime.h>
#include <hip/hip_bf16.h>

// ---------------------------------------------------------------------------
// LSTM forward for MI455X (gfx1250, wave32, WMMA bf16 -> fp32 accumulate)
//   T=512, B=64, IN=512, H=1024
// ---------------------------------------------------------------------------
#define T_STEPS 512
#define BATCH   64
#define IN_DIM  512
#define H_DIM   1024
#define G_DIM   (4 * H_DIM)      // 4096 gate columns
#define REC_BLOCKS 64            // persistent recurrent grid (j-tiles)
#define LDS_KSTRIDE 72           // 64 data elems + 8 pad (144B row: 36-dword lane
                                 // stride, near-conflict-free on 64 banks)

typedef __bf16 bf16;
typedef __attribute__((ext_vector_type(16))) __bf16 v16bf;
typedef __attribute__((ext_vector_type(8)))  __bf16 v8bf;
typedef __attribute__((ext_vector_type(8)))  float  v8f;
typedef __attribute__((ext_vector_type(4)))  unsigned int u32x4;
typedef __attribute__((ext_vector_type(8)))  int i32x8;
typedef __attribute__((ext_vector_type(4)))  int i32x4;

#if defined(__has_builtin)
#if __has_builtin(__builtin_amdgcn_tensor_load_to_lds) && \
    __has_builtin(__builtin_amdgcn_s_wait_tensorcnt)
#define USE_TDM 1
#endif
#endif
#ifndef USE_TDM
#define USE_TDM 0
#endif

// Build a 16-element bf16 fragment from two 16-byte chunks.
__device__ __forceinline__ v16bf load_frag16(const bf16* lo_p, const bf16* hi_p) {
    v8bf lo = *(const v8bf*)lo_p;
    v8bf hi = *(const v8bf*)hi_p;
    v16bf r;
#pragma unroll
    for (int i = 0; i < 8; ++i) { r[i] = lo[i]; r[i + 8] = hi[i]; }
    return r;
}

__device__ __forceinline__ float sigmoidf_fast(float x) {
    return 1.0f / (1.0f + __expf(-x));
}
__device__ __forceinline__ float tanhf_fast(float x) {
    // tanh(x) = 2*sigmoid(2x) - 1  -> single v_exp_f32
    return 2.0f / (1.0f + __expf(-2.0f * x)) - 1.0f;
}

// Sense-reversing grid-wide barrier: bar[0]=arrival count, bar[1]=generation.
__device__ __forceinline__ void grid_sync(unsigned* bar, unsigned nblocks) {
    __threadfence();
    __syncthreads();
    if (threadIdx.x == 0) {
        unsigned gen  = __hip_atomic_load(bar + 1, __ATOMIC_ACQUIRE, __HIP_MEMORY_SCOPE_AGENT);
        unsigned prev = __hip_atomic_fetch_add(bar, 1u, __ATOMIC_ACQ_REL, __HIP_MEMORY_SCOPE_AGENT);
        if (prev == nblocks - 1u) {
            __hip_atomic_store(bar, 0u, __ATOMIC_RELAXED, __HIP_MEMORY_SCOPE_AGENT);
            __hip_atomic_fetch_add(bar + 1, 1u, __ATOMIC_RELEASE, __HIP_MEMORY_SCOPE_AGENT);
        } else {
            while (__hip_atomic_load(bar + 1, __ATOMIC_ACQUIRE, __HIP_MEMORY_SCOPE_AGENT) == gen) {
                __builtin_amdgcn_s_sleep(2);
            }
        }
    }
    __syncthreads();
}

// ------------------------------- pack kernels ------------------------------
// Vectorized: 8 fp32 in (two float4) -> 8 bf16 out (one 16B store).
__global__ void lstm_pack_f32_to_bf16(const float* __restrict__ in,
                                      bf16* __restrict__ out, int n8) {
    for (int i = blockIdx.x * blockDim.x + threadIdx.x; i < n8;
         i += gridDim.x * blockDim.x) {
        const float4* p = (const float4*)in + (size_t)i * 2;
        float4 a = p[0], b = p[1];
        v8bf o;
        o[0] = (bf16)a.x; o[1] = (bf16)a.y; o[2] = (bf16)a.z; o[3] = (bf16)a.w;
        o[4] = (bf16)b.x; o[5] = (bf16)b.y; o[6] = (bf16)b.z; o[7] = (bf16)b.w;
        ((v8bf*)out)[i] = o;
    }
}

// in: [G, K] row-major fp32  ->  out: [K, G] row-major bf16 (coalesced writes)
__global__ void lstm_pack_transpose_bf16(const float* __restrict__ in,
                                         bf16* __restrict__ out, int G, int K) {
    int n = G * K;
    for (int i = blockIdx.x * blockDim.x + threadIdx.x; i < n;
         i += gridDim.x * blockDim.x) {
        int k = i / G;
        int g = i - k * G;
        out[i] = (bf16)in[(size_t)g * K + k];
    }
}

__global__ void lstm_bias_sum(const float* __restrict__ b_ih,
                              const float* __restrict__ b_hh,
                              float* __restrict__ bias, int n) {
    int i = blockIdx.x * blockDim.x + threadIdx.x;
    if (i < n) bias[i] = b_ih[i] + b_hh[i];
}

// ------------------------- x_proj GEMM (parallel in T) ---------------------
// xproj[32768, 4096] = Xbf[32768, 512] @ Wt_ih[512, 4096] + bias
// Block: 256 threads (8 waves). Wave = one N-tile, 4 consecutive M-tiles.
// Ping-pong software pipeline (fully unrolled -> no register copies).
__global__ void __launch_bounds__(256, 1)
lstm_xproj_gemm(const bf16* __restrict__ xbf,
                const bf16* __restrict__ wt_ih,   // [IN_DIM, G_DIM]
                const float* __restrict__ bias,
                float* __restrict__ xproj) {
    const int lane = threadIdx.x & 31;
    const int wave = threadIdx.x >> 5;
    const int r16  = lane & 15;      // A row within tile / D column
    const int hi16 = lane >> 4;      // k-half select / D row-half select

    const int m_base = blockIdx.x * 64;
    const int n0     = (blockIdx.y * 8 + wave) * 16;

    auto loadB = [&](int k0) {
        const bf16* bp = wt_ih + (size_t)(k0 + hi16 * 8) * G_DIM + n0 + r16;
        v16bf r;
#pragma unroll
        for (int i = 0; i < 8; ++i) { r[i] = bp[(size_t)i * G_DIM]; r[i + 8] = bp[(size_t)(i + 16) * G_DIM]; }
        return r;
    };
    auto loadA = [&](int mi, int k0) {
        const bf16* ap = xbf + (size_t)(m_base + mi * 16 + r16) * IN_DIM
                             + k0 + hi16 * 8;
        return load_frag16(ap, ap + 16);
    };

    v8f acc[4];
#pragma unroll
    for (int mi = 0; mi < 4; ++mi)
#pragma unroll
        for (int v = 0; v < 8; ++v) acc[mi][v] = 0.0f;

    v16bf bbuf[2];
    v16bf abuf[2][4];
    bbuf[0] = loadB(0);
#pragma unroll
    for (int mi = 0; mi < 4; ++mi) abuf[0][mi] = loadA(mi, 0);

#pragma unroll
    for (int kk = 0; kk < IN_DIM / 32; ++kk) {
        const int cur = kk & 1, nxt = cur ^ 1;
        if (kk < IN_DIM / 32 - 1) {           // prefetch next stage (other buffer)
            bbuf[nxt] = loadB((kk + 1) * 32);
#pragma unroll
            for (int mi = 0; mi < 4; ++mi) abuf[nxt][mi] = loadA(mi, (kk + 1) * 32);
        }
#pragma unroll
        for (int mi = 0; mi < 4; ++mi)
            acc[mi] = __builtin_amdgcn_wmma_f32_16x16x32_bf16(
                false, abuf[cur][mi], false, bbuf[cur], (short)0, acc[mi],
                false, false);
    }

    const float bv = bias[n0 + r16];
#pragma unroll
    for (int mi = 0; mi < 4; ++mi)
#pragma unroll
        for (int v = 0; v < 8; ++v)
            xproj[(size_t)(m_base + mi * 16 + v + 8 * hi16) * G_DIM + n0 + r16] =
                acc[mi][v] + bv;
}

// ------------------------- persistent recurrent kernel ---------------------
// 64 blocks x 128 threads (4 waves). block = j-tile (16 h-columns),
// wave = b-tile (16 batch rows). The block's W_hh slice (K=1024 x 64 gate
// columns) is staged ONCE into LDS (320KB/WGP on CDNA5) via the Tensor Data
// Mover (4 x 2D tile descriptors with LDS row padding), then reused by all
// 512 timesteps. Cell state c stays in registers for the whole sequence.
__global__ void __launch_bounds__(128, 1)
lstm_recurrent(const bf16* __restrict__ wt_hh,   // [H_DIM, G_DIM]
               const float* __restrict__ xproj,  // [T*B, G_DIM]
               bf16* __restrict__ hbf0,          // [BATCH, H_DIM] (step-even read)
               bf16* __restrict__ hbf1,          // [BATCH, H_DIM]
               float* __restrict__ out,          // [T,B,H] ++ hT[B,H] ++ cT[B,H]
               unsigned* __restrict__ bar) {
    __shared__ __align__(16) bf16 ldsw[H_DIM * LDS_KSTRIDE];   // 144KB

    const int lane = threadIdx.x & 31;
    const int wave = threadIdx.x >> 5;
    const int r16  = lane & 15;
    const int hi16 = lane >> 4;

    const int m0 = wave * 16;         // batch-row tile base
    const int j0 = blockIdx.x * 16;   // h-column tile base

#if USE_TDM
    // ---- TDM staging: 4 descriptors (one per gate panel, 16 x 1024 bf16).
    // Each row transfers 32B (8 dwords) then HW pads LDS by 28 dwords (112B)
    // -> 144B LDS row stride == LDS_KSTRIDE*2. Issued once from wave 0;
    // TENSORcnt fenced, then grid_sync publishes LDS to all waves.
    if (wave == 0) {
#pragma unroll
        for (int g = 0; g < 4; ++g) {
            unsigned long long gaddr =
                (unsigned long long)(size_t)(wt_hh + (size_t)g * H_DIM + j0);
            unsigned lds_addr = (unsigned)(size_t)(&ldsw[0]) + (unsigned)g * 32u;
            u32x4 d0;
            d0[0] = 1u;                                     // count=1 (valid D#)
            d0[1] = lds_addr;                               // LDS byte address
            d0[2] = (unsigned)(gaddr & 0xFFFFFFFFull);      // global addr lo
            d0[3] = (unsigned)((gaddr >> 32) & 0x1FFFFFFull)
                  | (2u << 30);                             // type=2 ("image")
            i32x8 d1;
            d1[0] = (1 << 16)        // data_size = 2 bytes
                  | (1 << 20)        // pad_enable
                  | (2 << 22)        // pad_interval: 8 dwords (=32B row payload)
                  | (27 << 25);      // pad_amount: 28 dwords (=112B LDS pad)
            d1[1] = (int)(4096u << 16);   // tensor_dim0[15:0]=4096 in bits[63:48]
            d1[2] = (int)(1024u << 16);   // tensor_dim0 hi=0; tensor_dim1=1024
            d1[3] = (16 << 16);           // tensor_dim1 hi=0; tile_dim0=16
            d1[4] = 1024;                 // tile_dim1=1024; tile_dim2=0
            d1[5] = 4096;                 // tensor_dim0_stride (elems) lo32
            d1[6] = 0;                    // stride hi; tensor_dim1_stride lo (2D: unused)
            d1[7] = 0;
            i32x4 z4 = {0, 0, 0, 0};      // groups 2/3 unused for 2D tiles
            i32x8 z8 = {0, 0, 0, 0, 0, 0, 0, 0};
            // 6-arg variant (clang-23 / therock-10.0 headers).
            __builtin_amdgcn_tensor_load_to_lds(d0, d1, z4, z4, z8, 0);
        }
        __builtin_amdgcn_s_wait_tensorcnt(0);
    }
#else
    // ---- Manual staging fallback (16B chunks, coalesced). ----
    for (int c = threadIdx.x; c < H_DIM * 8; c += blockDim.x) {
        int k = c >> 3, sub = c & 7, g = sub >> 1, half = sub & 1;
        const bf16* src = wt_hh + (size_t)k * G_DIM + g * H_DIM + j0 + half * 8;
        bf16*       dst = ldsw + k * LDS_KSTRIDE + g * 16 + half * 8;
        *(v8bf*)dst = *(const v8bf*)src;
    }
#endif

    // ---- Zero-init h(0) in global; c(0) lives in registers (this wave owns
    //      its 8 (b,j) elements exclusively for the entire sequence). ----
    float creg[8];
#pragma unroll
    for (int v = 0; v < 8; ++v) {
        creg[v] = 0.0f;
        hbf0[(m0 + v + 8 * hi16) * H_DIM + j0 + r16] = (bf16)0.0f;
    }
    grid_sync(bar, REC_BLOCKS);       // also covers the LDS fill (__syncthreads)

    for (int t = 0; t < T_STEPS; ++t) {
        const bf16* hread  = (t & 1) ? hbf1 : hbf0;
        bf16*       hwrite = (t & 1) ? hbf0 : hbf1;

        // Hint-prefetch next step's x_proj slice while we compute this one.
        if (t + 1 < T_STEPS) {
            const float* xpn = xproj + (size_t)((t + 1) * BATCH + m0) * G_DIM
                                     + j0 + r16 + (size_t)(8 * hi16) * G_DIM;
#pragma unroll
            for (int g = 0; g < 4; ++g) __builtin_prefetch(xpn + g * H_DIM, 0, 1);
        }

        // Seed accumulators with precomputed input projection (+ biases).
        v8f acc[4];
        const float* xp = xproj + (size_t)(t * BATCH + m0) * G_DIM + j0 + r16;
#pragma unroll
        for (int g = 0; g < 4; ++g)
#pragma unroll
            for (int v = 0; v < 8; ++v)
                acc[g][v] = xp[(size_t)(v + 8 * hi16) * G_DIM + g * H_DIM];

        // gates += h(t-1) @ W_hh^T  (K=1024; A from L2 ping-pong, B from LDS)
        auto loadA = [&](int k0) {
            const bf16* ap = hread + (size_t)(m0 + r16) * H_DIM + k0 + hi16 * 8;
            return load_frag16(ap, ap + 16);
        };
        v16bf abuf[2];
        abuf[0] = loadA(0);
#pragma unroll
        for (int kk = 0; kk < H_DIM / 32; ++kk) {
            const int cur = kk & 1, nxt = cur ^ 1;
            if (kk < H_DIM / 32 - 1) abuf[nxt] = loadA((kk + 1) * 32);
            const int k0 = kk * 32;
#pragma unroll
            for (int g = 0; g < 4; ++g) {
                const bf16* bp = ldsw + (k0 + hi16 * 8) * LDS_KSTRIDE + g * 16 + r16;
                v16bf bfrag;
#pragma unroll
                for (int i = 0; i < 8; ++i) { bfrag[i] = bp[i * LDS_KSTRIDE]; bfrag[i + 8] = bp[(i + 16) * LDS_KSTRIDE]; }
                acc[g] = __builtin_amdgcn_wmma_f32_16x16x32_bf16(
                    false, abuf[cur], false, bfrag, (short)0, acc[g],
                    false, false);
            }
        }

        // Lane-local LSTM cell update (C/D layout: M = v + 8*hi16, N = r16).
#pragma unroll
        for (int v = 0; v < 8; ++v) {
            const int b_row = m0 + v + 8 * hi16;
            const int idx   = b_row * H_DIM + j0 + r16;
            float ig = sigmoidf_fast(acc[0][v]);
            float fg = sigmoidf_fast(acc[1][v]);
            float gg = tanhf_fast(acc[2][v]);
            float og = sigmoidf_fast(acc[3][v]);
            float c_new = fg * creg[v] + ig * gg;
            float h_new = og * tanhf_fast(c_new);
            creg[v] = c_new;
            out[(size_t)t * (BATCH * H_DIM) + idx] = h_new;
            hwrite[idx] = (bf16)h_new;
            if (t == T_STEPS - 1) {
                out[(size_t)T_STEPS * (BATCH * H_DIM) + idx]                 = h_new;
                out[(size_t)T_STEPS * (BATCH * H_DIM) + BATCH * H_DIM + idx] = c_new;
            }
        }

        // Make h(t) visible device-wide before step t+1 reads it.
        grid_sync(bar, REC_BLOCKS);
    }
}

// --------------------------------- launch ----------------------------------
extern "C" void kernel_launch(void* const* d_in, const int* in_sizes, int n_in,
                              void* d_out, int out_size, void* d_ws, size_t ws_size,
                              hipStream_t stream) {
    const float* x    = (const float*)d_in[0];   // [T, B, IN]
    const float* w_ih = (const float*)d_in[1];   // [4H, IN]
    const float* b_ih = (const float*)d_in[2];   // [4H]
    const float* w_hh = (const float*)d_in[3];   // [4H, H]
    const float* b_hh = (const float*)d_in[4];   // [4H]
    float* out = (float*)d_out;

    // Workspace bump allocator (256B aligned).
    char*  ws  = (char*)d_ws;
    size_t off = 0;
    auto wsalloc = [&](size_t bytes) -> void* {
        void* p = ws + off;
        off = (off + bytes + 255) & ~(size_t)255;
        return p;
    };
    bf16*     xbf    = (bf16*)wsalloc((size_t)T_STEPS * BATCH * IN_DIM * 2);
    bf16*     wt_ih  = (bf16*)wsalloc((size_t)IN_DIM * G_DIM * 2);
    bf16*     wt_hh  = (bf16*)wsalloc((size_t)H_DIM * G_DIM * 2);
    float*    bias   = (float*)wsalloc((size_t)G_DIM * 4);
    float*    xproj  = (float*)wsalloc((size_t)T_STEPS * BATCH * G_DIM * 4);
    bf16*     hbf0   = (bf16*)wsalloc((size_t)BATCH * H_DIM * 2);
    bf16*     hbf1   = (bf16*)wsalloc((size_t)BATCH * H_DIM * 2);
    unsigned* bar    = (unsigned*)wsalloc(256);

    // Reset grid-barrier state (graph-capture safe).
    (void)hipMemsetAsync(bar, 0, 256, stream);

    // 1) Pack inputs/weights to WMMA-friendly bf16 layouts.
    lstm_pack_f32_to_bf16<<<4096, 256, 0, stream>>>(
        x, xbf, T_STEPS * BATCH * IN_DIM / 8);
    lstm_pack_transpose_bf16<<<2048, 256, 0, stream>>>(
        w_ih, wt_ih, G_DIM, IN_DIM);
    lstm_pack_transpose_bf16<<<4096, 256, 0, stream>>>(
        w_hh, wt_hh, G_DIM, H_DIM);
    lstm_bias_sum<<<(G_DIM + 255) / 256, 256, 0, stream>>>(
        b_ih, b_hh, bias, G_DIM);

    // 2) Big parallel GEMM: xproj = X @ W_ih^T + (b_ih + b_hh).
    dim3 ggrid(T_STEPS * BATCH / 64, G_DIM / (16 * 8));
    lstm_xproj_gemm<<<ggrid, 256, 0, stream>>>(xbf, wt_ih, bias, xproj);

    // 3) Persistent recurrent kernel: 512 steps with grid-wide barriers.
    lstm_recurrent<<<REC_BLOCKS, 128, 0, stream>>>(
        wt_hh, xproj, hbf0, hbf1, out, bar);

    (void)in_sizes; (void)n_in; (void)out_size; (void)ws_size;
}
